// VectorQuantizer_14096082665950
// MI455X (gfx1250) — matrix operand order repaired and checked by reading, not compiled, not run on hardware
//
#include <hip/hip_runtime.h>
#include <hip/hip_bf16.h>

typedef __attribute__((ext_vector_type(16))) __bf16 v16bf;
typedef __attribute__((ext_vector_type(8)))  __bf16 v8bf;
typedef __attribute__((ext_vector_type(8)))  float  v8f;

#define NUM_EMB 8192
#define EMB_DIM 512
#define NROWS   32768   // 16 * 2048
#define TROWS   64      // rows per block in the GEMM kernel

union V16U { v16bf v; v8bf h[2]; };

// ---------------------------------------------------------------------------
// Prep: split codebook fp32 -> bf16 hi/lo, compute ||w_k||^2 in fp32
// ---------------------------------------------------------------------------
__global__ __launch_bounds__(256)
void vq_prep_w(const float* __restrict__ w, __bf16* __restrict__ Wh,
               __bf16* __restrict__ Wl, float* __restrict__ wnorm) {
    __shared__ float red[256];
    const int k   = blockIdx.x;
    const int tid = threadIdx.x;
    float s = 0.f;
    for (int d = tid; d < EMB_DIM; d += 256) {
        float x = w[(long)k * EMB_DIM + d];
        __bf16 h = (__bf16)x;
        Wh[(long)k * EMB_DIM + d] = h;
        Wl[(long)k * EMB_DIM + d] = (__bf16)(x - (float)h);
        s += x * x;
    }
    red[tid] = s; __syncthreads();
    for (int st = 128; st > 0; st >>= 1) {
        if (tid < st) red[tid] += red[tid + st];
        __syncthreads();
    }
    if (tid == 0) wnorm[k] = red[0];
}

// ---------------------------------------------------------------------------
// Zero the scatter targets (counts lives in new_cs slot, dw in new_ema_w slot)
// ---------------------------------------------------------------------------
__global__ void vq_zero(float* __restrict__ counts, float* __restrict__ dw,
                        float* __restrict__ nacc) {
    long i = (long)blockIdx.x * blockDim.x + threadIdx.x;
    if (i < (long)NUM_EMB * EMB_DIM) dw[i] = 0.f;
    if (i < NUM_EMB) counts[i] = 0.f;
    if (i == 0) nacc[0] = 0.f;
}

// ---------------------------------------------------------------------------
// GEMM + fused argmin.
//   Block = 64 rows, 8 waves = 4 row-groups x 2 col-groups.
//   Per chunk of 32 codes, wave (rg,cg) computes a 16x16 tile:
//   rows rg*16.., codes chunk + cg*16.. ; the 4 row-group waves issue
//   identical B addresses -> WGP$ reuse, L2 traffic /4.
//   D=512 -> 16 k-steps of v_wmma_f32_16x16x32_bf16 x3 (bf16 hi/lo split).
//   Three accumulators (one per product term) so consecutive WMMAs never
//   chain through the same C/D registers.
//   score = ||w||^2 - 2*(z.w)   (||z||^2 constant per row -> dropped)
//
//   A LDS layout per ISA 16-bit A table (granule-interleaved):
//     lane<16 needs K granules {0,2}, lane>=16 needs {1,3} of each 32-chunk.
//     Stored as planes [kstep][granule][row][8 elems] -> each half-wave reads
//     16 consecutive 16B chunks = every LDS bank exactly once (conflict-free).
// ---------------------------------------------------------------------------
__global__ __launch_bounds__(256)
void vq_gemm_argmin(const float* __restrict__ z,
                    const __bf16* __restrict__ Wh,
                    const __bf16* __restrict__ Wl,
                    const float* __restrict__ wnorm,
                    int* __restrict__ codes) {
    __shared__ __bf16 sAh[16 * 4 * TROWS * 8];   // 64 KB
    __shared__ __bf16 sAl[16 * 4 * TROWS * 8];   // 64 KB
    __shared__ float  sMV[8][16];
    __shared__ int    sMI[8][16];

    const int tid  = threadIdx.x;
    const int lane = tid & 31;
    const int wv   = tid >> 5;        // wave 0..7
    const int rg   = wv >> 1;         // row group 0..3
    const int cg   = wv & 1;          // col group 0..1
    const int cr   = lane & 15;       // A-row / B-col owned by this lane
    const int half = lane >> 4;       // lane half
    const long rowBase = (long)blockIdx.x * TROWS;

    // stage 64x512 z tile: fp32 -> bf16 hi + lo, granule-plane layout
    for (int i = tid; i < TROWS * EMB_DIM; i += 256) {
        int r = i >> 9, d = i & 511;
        float x = z[(rowBase + r) * EMB_DIM + d];
        __bf16 h = (__bf16)x;
        __bf16 l = (__bf16)(x - (float)h);
        int s = d >> 5;            // 32-wide k-step
        int g = (d >> 3) & 3;      // 8-wide granule within k-step
        int e = d & 7;
        int idx = ((s * 4 + g) * TROWS + r) * 8 + e;
        sAh[idx] = h;
        sAl[idx] = l;
    }
    __syncthreads();

    const int aRow = rg * 16 + cr;    // this lane's A row within the tile

    float bestV[8];
    int   bestI[8];
#pragma unroll
    for (int i = 0; i < 8; i++) { bestV[i] = 3.402823e38f; bestI[i] = 0x7FFFFFFF; }

    for (int kc = 0; kc < NUM_EMB; kc += 32) {
        const int code = kc + cg * 16 + cr;          // this lane's B column
        const __bf16* wH = Wh + (long)code * EMB_DIM;
        const __bf16* wL = Wl + (long)code * EMB_DIM;
        v8f accHH = {};
        v8f accHL = {};
        v8f accLH = {};
#pragma unroll 4
        for (int s = 0; s < 16; s++) {
            // A fragment: granules {half, half+2} of this k-step
            V16U ah, al;
            ah.h[0] = *(const v8bf*)&sAh[((s * 4 + half)     * TROWS + aRow) * 8];
            ah.h[1] = *(const v8bf*)&sAh[((s * 4 + half + 2) * TROWS + aRow) * 8];
            al.h[0] = *(const v8bf*)&sAl[((s * 4 + half)     * TROWS + aRow) * 8];
            al.h[1] = *(const v8bf*)&sAl[((s * 4 + half + 2) * TROWS + aRow) * 8];
            // B fragment: granules {half, half+2} of this k-step (matches A)
            const int off = s * 32 + half * 8;
            V16U bhu, blu;
            bhu.h[0] = *(const v8bf*)(wH + off);
            bhu.h[1] = *(const v8bf*)(wH + off + 16);
            blu.h[0] = *(const v8bf*)(wL + off);
            blu.h[1] = *(const v8bf*)(wL + off + 16);
            v16bf bh = bhu.v;
            v16bf bl = blu.v;
            accHH = __builtin_amdgcn_wmma_f32_16x16x32_bf16(false, ah.v, false, bh,
                                                            (short)0, accHH, false, false);
            accHL = __builtin_amdgcn_wmma_f32_16x16x32_bf16(false, ah.v, false, bl,
                                                            (short)0, accHL, false, false);
            accLH = __builtin_amdgcn_wmma_f32_16x16x32_bf16(false, al.v, false, bh,
                                                            (short)0, accLH, false, false);
        }
        const float wn = wnorm[code];
#pragma unroll
        for (int i = 0; i < 8; i++) {
            float dot = accHH[i] + (accHL[i] + accLH[i]);
            float sc  = wn - 2.0f * dot;        // slot i -> tile row half*8+i, col cr
            if (sc < bestV[i] || (sc == bestV[i] && code < bestI[i])) {
                bestV[i] = sc; bestI[i] = code;
            }
        }
    }

    // reduce across the 16 lane-columns within each half-wave
#pragma unroll
    for (int m = 1; m < 16; m <<= 1) {
#pragma unroll
        for (int i = 0; i < 8; i++) {
            float ov = __shfl_xor(bestV[i], m, 32);
            int   oi = __shfl_xor(bestI[i], m, 32);
            if (ov < bestV[i] || (ov == bestV[i] && oi < bestI[i])) {
                bestV[i] = ov; bestI[i] = oi;
            }
        }
    }
    if (cr == 0) {
#pragma unroll
        for (int i = 0; i < 8; i++) {
            sMV[wv][half * 8 + i] = bestV[i];
            sMI[wv][half * 8 + i] = bestI[i];
        }
    }
    __syncthreads();
    if (tid < TROWS) {   // merge the 2 col-group waves of each row group
        const int mrg = tid >> 4, r16 = tid & 15;
        float bv = sMV[mrg * 2][r16];     int bi = sMI[mrg * 2][r16];
        float ov = sMV[mrg * 2 + 1][r16]; int oi = sMI[mrg * 2 + 1][r16];
        if (ov < bv || (ov == bv && oi < bi)) { bv = ov; bi = oi; }
        codes[rowBase + tid] = bi;
    }
}

// ---------------------------------------------------------------------------
// Segment sums: counts[c] += 1, dw[c,:] += z[n,:]
// ---------------------------------------------------------------------------
__global__ __launch_bounds__(256)
void vq_scatter(const float* __restrict__ z, const int* __restrict__ codes,
                float* __restrict__ counts, float* __restrict__ dw) {
    const int n   = blockIdx.x;
    const int c   = codes[n];
    const int tid = threadIdx.x;
    if (tid == 0) atomicAdd(&counts[c], 1.0f);
    for (int d = tid; d < EMB_DIM; d += 256)
        atomicAdd(&dw[(long)c * EMB_DIM + d], z[(long)n * EMB_DIM + d]);
}

// ---------------------------------------------------------------------------
// new_cs = 0.99*ema_cs + 0.01*counts (in place over counts); n = sum(new_cs)
// ---------------------------------------------------------------------------
__global__ __launch_bounds__(1024)
void vq_ema_cs(const float* __restrict__ ema_cs, float* __restrict__ newcs,
               float* __restrict__ nOut) {
    __shared__ float red[1024];
    const int tid = threadIdx.x;
    float s = 0.f;
    for (int k = tid; k < NUM_EMB; k += 1024) {
        float v = ema_cs[k] * 0.99f + 0.01f * newcs[k];
        newcs[k] = v;
        s += v;
    }
    red[tid] = s; __syncthreads();
    for (int st = 512; st > 0; st >>= 1) {
        if (tid < st) red[tid] += red[tid + st];
        __syncthreads();
    }
    if (tid == 0) nOut[0] = red[0];
}

// ---------------------------------------------------------------------------
// new_ema_w = 0.99*ema_w + 0.01*dw (in place over dw); new_weight = ./cs
// ---------------------------------------------------------------------------
__global__ __launch_bounds__(256)
void vq_ema_w(const float* __restrict__ ema_w, const float* __restrict__ newcs,
              const float* __restrict__ nPtr, float* __restrict__ new_ema_w,
              float* __restrict__ new_weight) {
    long i = (long)blockIdx.x * 256 + threadIdx.x;
    if (i >= (long)NUM_EMB * EMB_DIM) return;
    int k = (int)(i >> 9);
    float n  = nPtr[0];
    float cs = (newcs[k] + 1e-5f) / (n + (float)NUM_EMB * 1e-5f) * n;
    float w  = ema_w[i] * 0.99f + 0.01f * new_ema_w[i];
    new_ema_w[i]  = w;
    new_weight[i] = w / cs;
}

// ---------------------------------------------------------------------------
// z_q = new_weight[codes]; write z_q_out (straight-through fwd == z_q),
// codes_out (as float), per-row loss partials
// ---------------------------------------------------------------------------
__global__ __launch_bounds__(256)
void vq_gather(const float* __restrict__ z, const int* __restrict__ codes,
               const float* __restrict__ new_weight, float* __restrict__ zq_out,
               float* __restrict__ codes_out, float* __restrict__ part) {
    __shared__ float red[256];
    const int n   = blockIdx.x;
    const int c   = codes[n];
    const int tid = threadIdx.x;
    float s = 0.f;
    for (int d = tid; d < EMB_DIM; d += 256) {
        float q = new_weight[(long)c * EMB_DIM + d];
        float x = z[(long)n * EMB_DIM + d];
        zq_out[(long)n * EMB_DIM + d] = q;
        float df = q - x;
        s += df * df;
    }
    red[tid] = s; __syncthreads();
    for (int st = 128; st > 0; st >>= 1) {
        if (tid < st) red[tid] += red[tid + st];
        __syncthreads();
    }
    if (tid == 0) { part[n] = red[0]; codes_out[n] = (float)c; }
}

__global__ __launch_bounds__(256)
void vq_loss(const float* __restrict__ part, float* __restrict__ out) {
    __shared__ float red[256];
    const int tid = threadIdx.x;
    float s = 0.f;
    for (int i = tid; i < NROWS; i += 256) s += part[i];
    red[tid] = s; __syncthreads();
    for (int st = 128; st > 0; st >>= 1) {
        if (tid < st) red[tid] += red[tid + st];
        __syncthreads();
    }
    if (tid == 0) out[0] = 0.25f * red[0] / (float)((long)NROWS * EMB_DIM);
}

// ---------------------------------------------------------------------------
extern "C" void kernel_launch(void* const* d_in, const int* in_sizes, int n_in,
                              void* d_out, int out_size, void* d_ws, size_t ws_size,
                              hipStream_t stream) {
    const float* z      = (const float*)d_in[0];
    const float* weight = (const float*)d_in[1];
    const float* ema_cs = (const float*)d_in[2];
    const float* ema_w  = (const float*)d_in[3];

    // outputs concatenated: loss(1), z_q_out(N*D), codes(N), new_weight(K*D),
    // new_cs(K), new_ema_w(K*D)
    float* out      = (float*)d_out;
    float* o_loss   = out;
    float* o_zq     = out + 1;
    float* o_codes  = o_zq + (long)NROWS * EMB_DIM;
    float* o_weight = o_codes + NROWS;
    float* o_newcs  = o_weight + (long)NUM_EMB * EMB_DIM;
    float* o_emaw   = o_newcs + NUM_EMB;

    char* ws = (char*)d_ws;
    __bf16* Wh     = (__bf16*)(ws);                 // 8,388,608 B
    __bf16* Wl     = (__bf16*)(ws + 8388608);       // 8,388,608 B
    float*  wnorm  = (float*)(ws + 16777216);       //    32,768 B
    int*    codes  = (int*)(ws + 16809984);         //   131,072 B
    float*  nScal  = (float*)(ws + 16941056);       //       256 B
    float*  part   = (float*)(ws + 16941312);       //   131,072 B

    vq_prep_w<<<NUM_EMB, 256, 0, stream>>>(weight, Wh, Wl, wnorm);
    vq_zero<<<((long)NUM_EMB * EMB_DIM + 255) / 256, 256, 0, stream>>>(o_newcs, o_emaw, nScal);
    vq_gemm_argmin<<<NROWS / TROWS, 256, 0, stream>>>(z, Wh, Wl, wnorm, codes);
    vq_scatter<<<NROWS, 256, 0, stream>>>(z, codes, o_newcs, o_emaw);
    vq_ema_cs<<<1, 1024, 0, stream>>>(ema_cs, o_newcs, nScal);
    vq_ema_w<<<((long)NUM_EMB * EMB_DIM + 255) / 256, 256, 0, stream>>>(ema_w, o_newcs, nScal, o_emaw, o_weight);
    vq_gather<<<NROWS, 256, 0, stream>>>(z, codes, o_weight, o_zq, o_codes, part);
    vq_loss<<<1, 256, 0, stream>>>(part, o_loss);
}
